// parKT_73735998538193
// MI455X (gfx1250) — matrix operand order repaired and checked by reading, not compiled, not run on hardware
//
#include <hip/hip_runtime.h>

#define B_   64
#define S_   512
#define D_   512
#define H_   8
#define DK_  64
#define L_   2
#define DFF_ 1024
#define M_   (B_*S_)          // 32768 rows
#define NEGV (-1e32f)
#define SCALE_ 0.125f         // 1/sqrt(64)

typedef __attribute__((ext_vector_type(16))) _Float16 v16h;
typedef __attribute__((ext_vector_type(8)))  _Float16 v8h;
typedef __attribute__((ext_vector_type(8)))  float    v8f;

// ---------- CDNA5 async global->LDS copy (ASYNCcnt path, ISA 08 §4) ----------
__device__ __forceinline__ uint32_t to_lds(const void* p) {
    // generic LDS address: addr[31:0] is the LDS byte offset
    return (uint32_t)(uintptr_t)p;
}
__device__ __forceinline__ void async_ld_b128(uint32_t lds, const void* gaddr) {
    asm volatile("global_load_async_to_lds_b128 %0, %1, off"
                 :: "v"(lds), "v"((unsigned long long)(uintptr_t)gaddr)
                 : "memory");
}
__device__ __forceinline__ void wait_async0() {
    asm volatile("s_wait_asynccnt 0" ::: "memory");
}

// ---------- WMMA fragment loaders (16x16x32 f16; A and B both use contiguous k = 16*(lane/16) + s) ----------
// A (16x32, MxK), row-major source with leading dim ld
__device__ __forceinline__ v16h frag_A(const _Float16* src, int ld, int lane) {
    int r  = lane & 15;
    int k0 = (lane & 16) ? 16 : 0;
    const _Float16* p = src + r * ld + k0;
    v16h a;
#pragma unroll
    for (int i = 0; i < 8; ++i) a[i] = p[i];
#pragma unroll
    for (int i = 0; i < 8; ++i) a[8 + i] = p[8 + i];
    return a;
}
// B (32x16, KxN) where the tile is stored TRANSPOSED row-major [n][k]:
// per-lane fully contiguous 16 halfs -> two ds_load_b128
__device__ __forceinline__ v16h frag_Bt(const _Float16* src, int ld, int lane) {
    int c  = lane & 15;
    int kb = (lane & 16) ? 16 : 0;
    const _Float16* p = src + c * ld + kb;
    v16h b;
#pragma unroll
    for (int i = 0; i < 16; ++i) b[i] = p[i];
    return b;
}

#define WMMA_F16(a, b, c) \
    __builtin_amdgcn_wmma_f32_16x16x32_f16(false, (a), false, (b), (short)0, (c), false, false)

// ---------- prep: x = q + pe, y = qa + pe ----------
__global__ __launch_bounds__(256) void prep_kernel(
    const float* __restrict__ q, const float* __restrict__ qa,
    float* __restrict__ x, _Float16* __restrict__ xh, _Float16* __restrict__ yh) {
    size_t i = (size_t)blockIdx.x * 256 + threadIdx.x;   // < M_*D_
    int d = (int)(i & (D_ - 1));
    int s = (int)((i >> 9) & (S_ - 1));
    int j = d >> 1;
    float div = __expf((float)(2 * j) * -0.01798894561f); // -ln(10000)/512
    float ang = (float)s * div;
    float pe  = (d & 1) ? __cosf(ang) : __sinf(ang);
    float xv = q[i] + pe, yv = qa[i] + pe;
    x[i]  = xv;
    xh[i] = (_Float16)xv;
    yh[i] = (_Float16)yv;
}

// ---------- f32 -> f16 convert with transpose: dst[l][n][k] = src[l][k][n] ----------
__global__ __launch_bounds__(256) void cvt_t_kernel(
    const float* __restrict__ src, _Float16* __restrict__ dst, int K, int N, size_t total) {
    size_t i = (size_t)blockIdx.x * 256 + threadIdx.x;
    size_t stride = (size_t)gridDim.x * 256;
    for (; i < total; i += stride) {
        int k = (int)(i % K);
        size_t nk = i / K;
        int n = (int)(nk % N);
        int l = (int)(nk / N);
        dst[i] = (_Float16)src[((size_t)l * K + k) * N + n];
    }
}

// ---------- tiled WMMA GEMM: C[M,N] = A[M,K](f16) @ W(f16, pre-transposed [N,K]) + bias ----------
// block tile 128x128, 8 waves (4 Mwaves x 2 Nwaves), wave tile 32x64 = 2x4 frags
// K-step 64, double-buffered LDS (64KB) filled by global_load_async_to_lds_b128
template <bool RELU, bool OUT16>
__global__ __launch_bounds__(256) void gemm_f16(
    const _Float16* __restrict__ A, const _Float16* __restrict__ WT,
    const float* __restrict__ bias, float* __restrict__ Cf,
    _Float16* __restrict__ Ch, int M, int N, int K) {
    __shared__ _Float16 As[2][128 * 64];
    __shared__ _Float16 TBs[2][128 * 64];   // [n][k]
    int tid = threadIdx.x, lane = tid & 31, wave = tid >> 5;
    int wm = wave & 3, wn = wave >> 2;
    int bm = blockIdx.x * 128, bn = blockIdx.y * 128;

    v8f acc[2][4];
#pragma unroll
    for (int i = 0; i < 2; ++i)
#pragma unroll
        for (int j = 0; j < 4; ++j) acc[i][j] = (v8f)0.0f;

    auto stage = [&](int buf, int k0) {
#pragma unroll
        for (int c = 0; c < 4; ++c) {               // 128x64 halfs = 1024 16B chunks
            int chunk = tid * 4 + c;
            int r = chunk >> 3, col = (chunk & 7) * 8;
            async_ld_b128(to_lds(&As[buf][r * 64 + col]),
                          &A[(size_t)(bm + r) * K + k0 + col]);
            async_ld_b128(to_lds(&TBs[buf][r * 64 + col]),
                          &WT[(size_t)(bn + r) * K + k0 + col]);
        }
    };

    stage(0, 0);
    wait_async0();
    __syncthreads();

    int buf = 0;
    for (int k0 = 0; k0 < K; k0 += 64) {
        if (k0 + 64 < K) stage(buf ^ 1, k0 + 64);   // prefetch next tile while computing
#pragma unroll
        for (int kk = 0; kk < 64; kk += 32) {
            v16h af[2], bf[4];
#pragma unroll
            for (int i = 0; i < 2; ++i)
                af[i] = frag_A(&As[buf][(wm * 32 + i * 16) * 64 + kk], 64, lane);
#pragma unroll
            for (int j = 0; j < 4; ++j)
                bf[j] = frag_Bt(&TBs[buf][(wn * 64 + j * 16) * 64 + kk], 64, lane);
#pragma unroll
            for (int i = 0; i < 2; ++i)
#pragma unroll
                for (int j = 0; j < 4; ++j) acc[i][j] = WMMA_F16(af[i], bf[j], acc[i][j]);
        }
        wait_async0();
        __syncthreads();
        buf ^= 1;
    }

    int mhalf = (lane & 16) ? 8 : 0;
    int nl = lane & 15;
#pragma unroll
    for (int i = 0; i < 2; ++i)
#pragma unroll
        for (int j = 0; j < 4; ++j)
#pragma unroll
            for (int v = 0; v < 8; ++v) {
                int m = bm + wm * 32 + i * 16 + mhalf + v;
                int n = bn + wn * 64 + j * 16 + nl;
                float val = acc[i][j][v] + bias[n];
                if (RELU) val = fmaxf(val, 0.0f);
                if (OUT16) Ch[(size_t)m * N + n] = (_Float16)val;
                else       Cf[(size_t)m * N + n] = val;
            }
}

// ---------- flash attention (per b,h,64-row q tile; causal strict lower; forget_rate) ----------
// K tiles double-buffered via async-to-LDS; V prefetched to regs, transpose-stored;
// causal masking only applied on the (single, uniform-branch) diagonal tile
__global__ __launch_bounds__(128) void flash_attn_kernel(
    const _Float16* __restrict__ qk, const _Float16* __restrict__ vv,
    const float* __restrict__ forget, _Float16* __restrict__ out) {
    __shared__ _Float16 Qs[64 * 64];
    __shared__ _Float16 Ks[2][64 * 64];     // [key][d]
    __shared__ _Float16 Vt[2][64 * 64];     // [d][key] (transposed)
    __shared__ _Float16 Ps[4][16 * 64];

    int bh = blockIdx.x, b = bh / H_, h = bh % H_;
    int qt = blockIdx.y;
    int tid = threadIdx.x, lane = tid & 31, wave = tid >> 5;
    const _Float16* qbase = qk + (size_t)b * S_ * D_ + h * DK_;
    const _Float16* vbase = vv + (size_t)b * S_ * D_ + h * DK_;

    // chunk geometry (shared by Q/K/V tile staging): 64x64 halfs, 512 chunks, 4/thread
    int cr[4], cc[4];
#pragma unroll
    for (int c = 0; c < 4; ++c) {
        int chunk = tid * 4 + c;
        cr[c] = chunk >> 3;
        cc[c] = (chunk & 7) * 8;
    }

#pragma unroll
    for (int c = 0; c < 4; ++c)                     // Q tile
        *(v8h*)&Qs[cr[c] * 64 + cc[c]] =
            *(const v8h*)&qbase[(size_t)(qt * 64 + cr[c]) * D_ + cc[c]];

    v8h vreg[4];
#pragma unroll
    for (int c = 0; c < 4; ++c) {                   // async K0, V0 -> regs
        async_ld_b128(to_lds(&Ks[0][cr[c] * 64 + cc[c]]),
                      &qbase[(size_t)cr[c] * D_ + cc[c]]);
        vreg[c] = *(const v8h*)&vbase[(size_t)cr[c] * D_ + cc[c]];
    }
    wait_async0();
#pragma unroll
    for (int c = 0; c < 4; ++c)                     // V0 transpose-store
#pragma unroll
        for (int j = 0; j < 8; ++j) Vt[0][(cc[c] + j) * 64 + cr[c]] = vreg[c][j];
    __syncthreads();

    int mhalf = (lane & 16) ? 8 : 0;
    int nl = lane & 15;
    int qrow_base = qt * 64 + wave * 16;
    int lmb = wave * 16 + mhalf;                    // local row base within 64-row q tile

    // Q fragments are loop-invariant: hoist out of the kt loop
    v16h aq[2];
#pragma unroll
    for (int kk = 0; kk < 2; ++kk)
        aq[kk] = frag_A(&Qs[(wave * 16) * 64 + kk * 32], 64, lane);

    float frg[8], rmax[8], rsum[8];
    v8f accO[4];
#pragma unroll
    for (int v = 0; v < 8; ++v) {
        rmax[v] = -INFINITY;
        rsum[v] = 0.0f;
        frg[v]  = forget[(size_t)b * S_ + qrow_base + mhalf + v] * SCALE_;
    }
#pragma unroll
    for (int j = 0; j < 4; ++j) accO[j] = (v8f)0.0f;

    int buf = 0;
    for (int kt = 0; kt <= qt; ++kt) {
        if (kt < qt) {                              // prefetch next K (async) and V (regs)
#pragma unroll
            for (int c = 0; c < 4; ++c) {
                async_ld_b128(to_lds(&Ks[buf ^ 1][cr[c] * 64 + cc[c]]),
                              &qbase[(size_t)((kt + 1) * 64 + cr[c]) * D_ + cc[c]]);
                vreg[c] = *(const v8h*)&vbase[(size_t)((kt + 1) * 64 + cr[c]) * D_ + cc[c]];
            }
        }

        v8f sc[4];
#pragma unroll
        for (int nf = 0; nf < 4; ++nf) sc[nf] = (v8f)0.0f;
#pragma unroll
        for (int kk = 0; kk < 2; ++kk)              // S = Q @ K^T
#pragma unroll
            for (int nf = 0; nf < 4; ++nf) {
                v16h bfr = frag_Bt(&Ks[buf][(nf * 16) * 64 + kk * 32], 64, lane);
                sc[nf] = WMMA_F16(aq[kk], bfr, sc[nf]);
            }

        // scale by forget_rate / sqrt(dk)
#pragma unroll
        for (int nf = 0; nf < 4; ++nf)
#pragma unroll
            for (int v = 0; v < 8; ++v) sc[nf][v] *= frg[v];

        if (kt == qt) {                             // causal mask: only the diagonal tile
#pragma unroll
            for (int nf = 0; nf < 4; ++nf)
#pragma unroll
                for (int v = 0; v < 8; ++v)
                    if (nf * 16 + nl >= lmb + v) sc[nf][v] = NEGV;  // allow only k < q
        }

        float tmax[8], tsum[8];
#pragma unroll
        for (int v = 0; v < 8; ++v) {
            tmax[v] = fmaxf(fmaxf(sc[0][v], sc[1][v]), fmaxf(sc[2][v], sc[3][v]));
            tsum[v] = 0.0f;
        }
#pragma unroll
        for (int off = 1; off < 16; off <<= 1)
#pragma unroll
            for (int v = 0; v < 8; ++v)
                tmax[v] = fmaxf(tmax[v], __shfl_xor(tmax[v], off, 32));

        float corr[8];
#pragma unroll
        for (int v = 0; v < 8; ++v) {
            float nm = fmaxf(rmax[v], tmax[v]);
            corr[v] = __expf(rmax[v] - nm);
            rmax[v] = nm;
        }
#pragma unroll
        for (int nf = 0; nf < 4; ++nf)
#pragma unroll
            for (int v = 0; v < 8; ++v) {
                float p = __expf(sc[nf][v] - rmax[v]);
                sc[nf][v] = p;
                tsum[v] += p;
            }
#pragma unroll
        for (int off = 1; off < 16; off <<= 1)
#pragma unroll
            for (int v = 0; v < 8; ++v) tsum[v] += __shfl_xor(tsum[v], off, 32);
#pragma unroll
        for (int v = 0; v < 8; ++v) rsum[v] = rsum[v] * corr[v] + tsum[v];
#pragma unroll
        for (int nf = 0; nf < 4; ++nf)
#pragma unroll
            for (int v = 0; v < 8; ++v) accO[nf][v] *= corr[v];

#pragma unroll
        for (int nf = 0; nf < 4; ++nf)              // stage P to LDS in f16
#pragma unroll
            for (int v = 0; v < 8; ++v)
                Ps[wave][(mhalf + v) * 64 + nf * 16 + nl] = (_Float16)sc[nf][v];

#pragma unroll
        for (int kk = 0; kk < 2; ++kk) {            // O += P @ V  (V stored transposed)
            v16h a = frag_A(&Ps[wave][kk * 32], 64, lane);
#pragma unroll
            for (int nf = 0; nf < 4; ++nf) {
                v16h bfr = frag_Bt(&Vt[buf][(nf * 16) * 64 + kk * 32], 64, lane);
                accO[nf] = WMMA_F16(a, bfr, accO[nf]);
            }
        }

        if (kt < qt) {                              // commit prefetched V (transposed)
#pragma unroll
            for (int c = 0; c < 4; ++c)
#pragma unroll
                for (int j = 0; j < 8; ++j) Vt[buf ^ 1][(cc[c] + j) * 64 + cr[c]] = vreg[c][j];
        }
        wait_async0();
        __syncthreads();
        buf ^= 1;
    }

#pragma unroll
    for (int nf = 0; nf < 4; ++nf)
#pragma unroll
        for (int v = 0; v < 8; ++v) {
            int qrow = qrow_base + mhalf + v;
            float o = accO[nf][v] / rsum[v];
            if (qrow == 0) o = 0.0f;                 // zero_pad first query row
            out[((size_t)b * S_ + qrow) * D_ + h * DK_ + nf * 16 + nl] = (_Float16)o;
        }
}

// ---------- residual + layernorm, writes f32 master + f16 copy ----------
__global__ __launch_bounds__(256) void ln_kernel(
    float* __restrict__ x, const float* __restrict__ t,
    const float* __restrict__ g, const float* __restrict__ bta,
    _Float16* __restrict__ xh) {
    __shared__ float red[16];
    size_t off = (size_t)blockIdx.x * D_;
    int tid = threadIdx.x;
    float v0 = x[off + tid] + t[off + tid];
    float v1 = x[off + tid + 256] + t[off + tid + 256];
    float s = v0 + v1, sq = v0 * v0 + v1 * v1;
#pragma unroll
    for (int o = 1; o < 32; o <<= 1) {
        s  += __shfl_xor(s,  o, 32);
        sq += __shfl_xor(sq, o, 32);
    }
    if ((tid & 31) == 0) { red[tid >> 5] = s; red[8 + (tid >> 5)] = sq; }
    __syncthreads();
    float ts = 0.0f, tq = 0.0f;
#pragma unroll
    for (int i = 0; i < 8; ++i) { ts += red[i]; tq += red[8 + i]; }
    float mean = ts * (1.0f / D_);
    float var  = tq * (1.0f / D_) - mean * mean;
    float inv  = rsqrtf(var + 1e-5f);
    float o0 = (v0 - mean) * inv * g[tid]       + bta[tid];
    float o1 = (v1 - mean) * inv * g[tid + 256] + bta[tid + 256];
    x[off + tid]        = o0;
    x[off + tid + 256]  = o1;
    xh[off + tid]       = (_Float16)o0;
    xh[off + tid + 256] = (_Float16)o1;
}

// ---------- orchestration ----------
extern "C" void kernel_launch(void* const* d_in, const int* in_sizes, int n_in,
                              void* d_out, int out_size, void* d_ws, size_t ws_size,
                              hipStream_t stream) {
    (void)in_sizes; (void)n_in; (void)out_size; (void)ws_size;
    const float* q      = (const float*)d_in[0];
    const float* qa     = (const float*)d_in[1];
    const float* forget = (const float*)d_in[2];
    const float* Wk = (const float*)d_in[3];
    const float* bk = (const float*)d_in[4];
    const float* Wv = (const float*)d_in[5];
    const float* bv = (const float*)d_in[6];
    const float* Wo = (const float*)d_in[7];
    const float* bo = (const float*)d_in[8];
    const float* ln1g = (const float*)d_in[9];
    const float* ln1b = (const float*)d_in[10];
    const float* W1 = (const float*)d_in[11];
    const float* b1 = (const float*)d_in[12];
    const float* W2 = (const float*)d_in[13];
    const float* b2 = (const float*)d_in[14];
    const float* ln2g = (const float*)d_in[15];
    const float* ln2b = (const float*)d_in[16];
    float* x = (float*)d_out;

    char* w = (char*)d_ws;
    auto alloc_h = [&](size_t n) { _Float16* p = (_Float16*)w; w += n * sizeof(_Float16); return p; };
    _Float16* xh    = alloc_h((size_t)M_ * D_);
    _Float16* yh    = alloc_h((size_t)M_ * D_);
    _Float16* attnh = alloc_h((size_t)M_ * D_);
    _Float16* qkh   = alloc_h((size_t)M_ * D_);
    _Float16* vh    = alloc_h((size_t)M_ * D_);
    _Float16* ffh   = qkh;   // FFN hidden (M x DFF) aliases qkh+vh after attention
    float* tmp = (float*)w;  w += (size_t)M_ * D_ * sizeof(float);
    _Float16* wkh = alloc_h((size_t)L_ * D_ * D_);     // transposed [N][K]
    _Float16* wvh = alloc_h((size_t)L_ * D_ * D_);
    _Float16* woh = alloc_h((size_t)L_ * D_ * D_);
    _Float16* w1h = alloc_h((size_t)L_ * DFF_ * D_);   // W1^T: [DFF][D]
    _Float16* w2h = alloc_h((size_t)L_ * D_ * DFF_);   // W2^T: [D][DFF]

    prep_kernel<<<(M_ * D_) / 256, 256, 0, stream>>>(q, qa, x, xh, yh);
    cvt_t_kernel<<<2048, 256, 0, stream>>>(Wk, wkh, D_, D_,  (size_t)L_ * D_ * D_);
    cvt_t_kernel<<<2048, 256, 0, stream>>>(Wv, wvh, D_, D_,  (size_t)L_ * D_ * D_);
    cvt_t_kernel<<<2048, 256, 0, stream>>>(Wo, woh, D_, D_,  (size_t)L_ * D_ * D_);
    cvt_t_kernel<<<4096, 256, 0, stream>>>(W1, w1h, D_, DFF_, (size_t)L_ * D_ * DFF_);
    cvt_t_kernel<<<4096, 256, 0, stream>>>(W2, w2h, DFF_, D_, (size_t)L_ * DFF_ * D_);

    for (int l = 0; l < L_; ++l) {
        gemm_f16<false, true><<<dim3(M_ / 128, D_ / 128), 256, 0, stream>>>(
            xh, wkh + (size_t)l * D_ * D_, bk + l * D_, nullptr, qkh, M_, D_, D_);
        gemm_f16<false, true><<<dim3(M_ / 128, D_ / 128), 256, 0, stream>>>(
            yh, wvh + (size_t)l * D_ * D_, bv + l * D_, nullptr, vh, M_, D_, D_);
        flash_attn_kernel<<<dim3(B_ * H_, S_ / 64), 128, 0, stream>>>(qkh, vh, forget, attnh);
        gemm_f16<false, false><<<dim3(M_ / 128, D_ / 128), 256, 0, stream>>>(
            attnh, woh + (size_t)l * D_ * D_, bo + l * D_, tmp, nullptr, M_, D_, D_);
        ln_kernel<<<M_, 256, 0, stream>>>(x, tmp, ln1g + l * D_, ln1b + l * D_, xh);
        gemm_f16<true, true><<<dim3(M_ / 128, DFF_ / 128), 256, 0, stream>>>(
            xh, w1h + (size_t)l * DFF_ * D_, b1 + l * DFF_, nullptr, ffh, M_, DFF_, D_);
        gemm_f16<false, false><<<dim3(M_ / 128, D_ / 128), 256, 0, stream>>>(
            ffh, w2h + (size_t)l * D_ * DFF_, b2 + l * D_, tmp, nullptr, M_, D_, DFF_);
        ln_kernel<<<M_, 256, 0, stream>>>(x, tmp, ln2g + l * D_, ln2b + l * D_, xh);
    }
}
